// GaussianLSS_48584670052372
// MI455X (gfx1250) — compile-verified, matrix operand order checked by reading, not run on hardware
//
#include <hip/hip_runtime.h>
#include <cstdint>

typedef __attribute__((ext_vector_type(16))) _Float16 v16h;
typedef __attribute__((ext_vector_type(8)))  _Float16 v8h;
typedef __attribute__((ext_vector_type(2)))  _Float16 v2h;
typedef __attribute__((ext_vector_type(8)))  float    v8f;

// ---------------- problem constants ----------------
constexpr int BN    = 12;     // b*n camera items
constexpr int H     = 32;
constexpr int W     = 88;
constexpr int NPIX  = H * W;  // 2816 gaussians per item
constexpr int DNUM  = 64;
constexpr int CH    = 128;
constexpr int GRID  = 200;
constexpr int TDIM  = 16;                 // tile edge
constexpr int TPS   = 13;                 // tiles per side (ceil(200/16))
constexpr int TILES = TPS * TPS;          // 169
constexpr int NTILES = BN * TILES;        // 2028
constexpr float ALPHA_MIN = 1.0f / 255.0f;

// ---------------- workspace layout (in 4-byte units) ----------------
constexpr int OFF_EXT  = 0;                       // 12 * 12 floats (3x4 inverse per item)
constexpr int OFF_PAR  = 160;                     // BN*NPIX*8 floats
constexpr int OFF_CNT  = OFF_PAR + BN * NPIX * 8; // ints from here
constexpr int OFF_OFFS = OFF_CNT  + 2048;         // NTILES+1 ints (exclusive scan)
constexpr int OFF_CUR  = OFF_OFFS + 2080;         // NTILES ints (fill cursors)
constexpr int OFF_LIST = OFF_CUR  + 2048;         // up to 4*BN*NPIX ints

// ================= rigid 4x4 inverse: [R|t] -> [R^T | -R^T t] =================
__global__ void ext_inv_kernel(const float* __restrict__ ext, float* __restrict__ extinv) {
    int b = threadIdx.x;
    if (b >= BN) return;
    const float* E = ext + b * 16;
    float R[3][3], t[3];
    for (int i = 0; i < 3; i++) {
        for (int j = 0; j < 3; j++) R[i][j] = E[i * 4 + j];
        t[i] = E[i * 4 + 3];
    }
    float* M = extinv + b * 12;
    for (int i = 0; i < 3; i++) {
        M[i * 4 + 0] = R[0][i];
        M[i * 4 + 1] = R[1][i];
        M[i * 4 + 2] = R[2][i];
        M[i * 4 + 3] = -(R[0][i] * t[0] + R[1][i] * t[1] + R[2][i] * t[2]);
    }
}

// ================= zero the tile counters =================
__global__ void zero_cnt_kernel(int* __restrict__ cnt) {
    int i = blockIdx.x * blockDim.x + threadIdx.x;
    if (i < NTILES) cnt[i] = 0;
}

// ================= per-gaussian prep: softmax moments -> conic; tile counting =================
__global__ __launch_bounds__(256)
void prep_kernel(const float* __restrict__ angles, const float* __restrict__ depth,
                 const float* __restrict__ opac, const float* __restrict__ extinv,
                 float* __restrict__ params, int* __restrict__ tileCnt) {
    int gid = blockIdx.x * blockDim.x + threadIdx.x;
    if (gid >= BN * NPIX) return;
    int b = gid / NPIX;
    int n = gid - b * NPIX;

    // softmax moments over depth bins: coords(d) = dbin*v + w  (affine in dbin)
    const float* dp = depth + (size_t)b * DNUM * NPIX + n;
    float x[DNUM];
    float mx = -1e30f;
#pragma unroll
    for (int d = 0; d < DNUM; d++) { x[d] = dp[(size_t)d * NPIX]; mx = fmaxf(mx, x[d]); }
    float s0 = 0.f, s1 = 0.f, s2 = 0.f;
#pragma unroll
    for (int d = 0; d < DNUM; d++) {
        float e  = __expf(x[d] - mx);
        float db = 1.0f + 0.9375f * (float)d;      // DEPTH_START + (60/64)*d
        s0 += e; s1 += e * db; s2 += e * db * db;
    }
    float inv  = 1.f / s0;
    float mean = s1 * inv;
    float var  = fmaxf(s2 * inv - mean * mean, 0.f);

    float th = angles[(size_t)gid * 2 + 0];
    float ph = angles[(size_t)gid * 2 + 1];
    float sth, cth, sph, cph;
    __sincosf(th, &sth, &cth);
    __sincosf(ph, &sph, &cph);
    float ux = sph * sth, uy = sph * cth, uz = cph;

    const float* M = extinv + b * 12;
    float vx = M[0] * ux + M[1] * uy + M[2] * uz;
    float vy = M[4] * ux + M[5] * uy + M[6] * uz;
    float px = mean * vx + M[3];
    float py = mean * vy + M[7];

    float row = -2.f * py + 100.f;   // -SH*y + GRID/2
    float col = -2.f * px + 100.f;   // -SW*x + GRID/2

    float sc  = var * (1.f / 9.f);   // ERR_TOL^2/9, rank-1 cov = sc * v v^T
    float Cxx = sc * vx * vx, Cxy = sc * vx * vy, Cyy = sc * vy * vy;
    float a  = 4.f * Cyy + 0.3f;     // SH^2*Cyy + LOWPASS
    float bb = 4.f * Cxy;
    float c  = 4.f * Cxx + 0.3f;
    float det = fmaxf(a * c - bb * bb, 1e-6f);
    float id  = 1.f / det;

    float op    = opac[gid];
    float opEff = (op > 0.05f) ? op : 0.f;       // valid mask folded into opacity
    float fr = floorf(row), fc = floorf(col);

    float* P = params + (size_t)gid * 8;
    P[0] = row; P[1] = col; P[2] = c * id; P[3] = -bb * id;
    P[4] = a * id; P[5] = opEff; P[6] = fr; P[7] = fc;

    if (opEff > 0.f && fr >= -4.f && fr <= 203.f && fc >= -4.f && fc <= 203.f) {
        int r0 = max((int)fr - 4, 0), r1 = min((int)fr + 4, GRID - 1);
        int c0 = max((int)fc - 4, 0), c1 = min((int)fc + 4, GRID - 1);
        if (r0 <= r1 && c0 <= c1) {
            for (int tr = r0 >> 4; tr <= (r1 >> 4); tr++)
                for (int tc = c0 >> 4; tc <= (c1 >> 4); tc++)
                    atomicAdd(&tileCnt[b * TILES + tr * TPS + tc], 1);
        }
    }
}

// ================= exclusive scan over 2028 tiles (trivial size) =================
__global__ void scan_kernel(const int* __restrict__ cnt, int* __restrict__ offs,
                            int* __restrict__ cur) {
    if (blockIdx.x == 0 && threadIdx.x == 0) {
        int acc = 0;
        for (int i = 0; i < NTILES; i++) { offs[i] = acc; cur[i] = acc; acc += cnt[i]; }
        offs[NTILES] = acc;
    }
}

// ================= fill per-tile gaussian lists =================
__global__ __launch_bounds__(256)
void fill_kernel(const float* __restrict__ params, int* __restrict__ cur,
                 int* __restrict__ list) {
    int gid = blockIdx.x * blockDim.x + threadIdx.x;
    if (gid >= BN * NPIX) return;
    int b = gid / NPIX;
    int n = gid - b * NPIX;
    const float* P = params + (size_t)gid * 8;
    float opEff = P[5], fr = P[6], fc = P[7];
    if (!(opEff > 0.f && fr >= -4.f && fr <= 203.f && fc >= -4.f && fc <= 203.f)) return;
    int r0 = max((int)fr - 4, 0), r1 = min((int)fr + 4, GRID - 1);
    int c0 = max((int)fc - 4, 0), c1 = min((int)fc + 4, GRID - 1);
    if (r0 > r1 || c0 > c1) return;
    for (int tr = r0 >> 4; tr <= (r1 >> 4); tr++)
        for (int tc = c0 >> 4; tc <= (c1 >> 4); tc++) {
            int slot = atomicAdd(&cur[b * TILES + tr * TPS + tc], 1);
            list[slot] = n;
        }
}

// ================= WMMA tile splat =================
// One workgroup = one (batch, 16x16 grid tile). 16 waves; wave w owns grid row
// TR*16+w. D[col(16), ch(16)] += A[col, 32 gauss] x B[32 gauss, ch] via
// v_wmma_f32_16x16x32_f16. Colors staged transposed in LDS (B fragment = two
// ds_load_b128); global gathers for chunk n+1 overlap chunk n's WMMAs
// (register double-buffer); B fragments software-pipelined so the LDS wait
// for fragment q+1 retires under WMMA q.
__global__ __launch_bounds__(512, 1)
void splat_kernel(const float* __restrict__ features, const float* __restrict__ params,
                  const int* __restrict__ tileOff, const int* __restrict__ tileList,
                  float* __restrict__ out) {
    __shared__ __align__(16) float    sP[32][8];     // gaussian params
    __shared__ __align__(16) _Float16 sColT[CH][32]; // colors TRANSPOSED: [channel][gauss]

    const int tileId = blockIdx.x;                 // b*169 + tr*13 + tc
    const int b  = tileId / TILES;
    const int t  = tileId - b * TILES;
    const int TR = t / TPS, TC = t - TR * TPS;

    const int tid  = threadIdx.x;
    const int wv   = tid >> 5;        // wave id 0..15 -> tile row
    const int lane = tid & 31;
    const int half = lane >> 4;       // K-half selector for 16-bit WMMA layout
    const int m    = lane & 15;       // A: M index (tile col); B: N index (channel)

    const int   r    = TR * TDIM + wv;
    const int   ccol = TC * TDIM + m;
    const float rf = (float)r, cf = (float)ccol;

    const int off = tileOff[tileId];
    const int cnt = tileOff[tileId + 1] - off;

    // staging assignment: thread owns gaussian pair g0,g0+1 across 4 channel groups
    const int g0  = (tid & 15) * 2;
    const int chb = tid >> 4;         // 0..31; ch_i = i*32 + chb

    v8f acc[8];
#pragma unroll
    for (int q = 0; q < 8; q++)
#pragma unroll
        for (int j = 0; j < 8; j++) acc[q][j] = 0.f;

    // ---- register staging buffers (next chunk) ----
    float fv0[4], fv1[4];             // features for (ch_i, g0), (ch_i, g0+1)
    float pr[8];                      // params (threads < 32)

    auto load_stage = [&](int base) {
        int i0 = base + g0, i1 = i0 + 1;
        int pix0 = (i0 < cnt) ? tileList[off + i0] : 0;
        int pix1 = (i1 < cnt) ? tileList[off + i1] : 0;
        const float* fb = features + (size_t)b * CH * NPIX;
#pragma unroll
        for (int i = 0; i < 4; i++) {
            int ch = i * 32 + chb;
            fv0[i] = fb[(size_t)ch * NPIX + pix0];
            fv1[i] = fb[(size_t)ch * NPIX + pix1];
        }
        if (tid < 32) {
            int idx = base + tid;
            int pix = (idx < cnt) ? tileList[off + idx] : -1;
            int p0  = (pix < 0) ? 0 : pix;
            const float4* P = (const float4*)(params + (size_t)(b * NPIX + p0) * 8);
            float4 a0 = P[0], a1 = P[1];
            pr[0] = a0.x; pr[1] = a0.y; pr[2] = a0.z; pr[3] = a0.w;
            pr[4] = a1.x; pr[5] = (pix < 0) ? 0.f : a1.y; pr[6] = a1.z; pr[7] = a1.w;
        }
        if (tid == 0)
            __builtin_prefetch(&tileList[off + base + 32], 0, 1);  // global_prefetch_b8
    };

    auto store_stage = [&]() {
        if (tid < 32) {
            *(float4*)&sP[tid][0] = make_float4(pr[0], pr[1], pr[2], pr[3]);
            *(float4*)&sP[tid][4] = make_float4(pr[4], pr[5], pr[6], pr[7]);
        }
#pragma unroll
        for (int i = 0; i < 4; i++) {
            int ch = i * 32 + chb;
            v2h pk; pk[0] = (_Float16)fv0[i]; pk[1] = (_Float16)fv1[i];
            *(v2h*)&sColT[ch][g0] = pk;       // packed b32 DS store
        }
    };

    auto loadB = [&](int q) -> v16h {
        int ch = q * 16 + m;
        v8h lo = *(const v8h*)&sColT[ch][half * 8];
        v8h hi = *(const v8h*)&sColT[ch][16 + half * 8];
        return __builtin_shufflevector(lo, hi, 0, 1, 2, 3, 4, 5, 6, 7,
                                       8, 9, 10, 11, 12, 13, 14, 15);
    };

    if (cnt > 0) load_stage(0);
    for (int base = 0; base < cnt; base += 32) {
        __syncthreads();               // previous chunk's LDS fully consumed
        store_stage();
        __syncthreads();               // LDS ready
        if (base + 32 < cnt) load_stage(base + 32);  // overlap gathers with WMMAs

        // A fragment: alphas for this lane's grid cell vs its 16 K-slots
        v16h A;
#pragma unroll
        for (int i = 0; i < 16; i++) {
            int k = (i < 8) ? (half * 8 + i) : (16 + half * 8 + (i - 8));
            float4 p0 = *(const float4*)&sP[k][0];   // row, col, cA, cB
            float4 p1 = *(const float4*)&sP[k][4];   // cC, op, fr, fc
            float dr = rf - p0.x, dc = cf - p0.y;
            float pw = fminf(-0.5f * (p0.z * dr * dr + p1.x * dc * dc) - p0.w * dr * dc, 0.f);
            float al = fminf(p1.y * __expf(pw), 0.99f);
            bool ok = (fabsf(rf - p1.z) <= 4.0f) && (fabsf(cf - p1.w) <= 4.0f) &&
                      (r < GRID) && (ccol < GRID) && (al >= ALPHA_MIN);
            A[i] = (_Float16)(ok ? al : 0.f);
        }
        // 8 channel chunks -> 8 WMMAs, B loads software-pipelined one deep
        v16h Bv = loadB(0);
#pragma unroll
        for (int q = 0; q < 8; q++) {
            v16h Bn = Bv;
            if (q < 7) Bn = loadB(q + 1);             // issue next loads first
            acc[q] = __builtin_amdgcn_wmma_f32_16x16x32_f16(
                false, A, false, Bv, (short)0, acc[q], false, false);
            Bv = Bn;
        }
    }

    // store: D layout -> acc[q][j] = value at (col = TC*16 + j + half*8, ch = q*16 + m)
    // tiles are disjoint: plain streaming stores, every output element covered once.
    const int cbase = TC * TDIM + half * 8;
    if (r < GRID) {
#pragma unroll
        for (int q = 0; q < 8; q++) {
            int ch = q * 16 + m;
            size_t rowbase = ((size_t)(b * CH + ch) * GRID + r) * GRID;
            if (cbase + 3 < GRID) {
                float4 v0 = make_float4(acc[q][0], acc[q][1], acc[q][2], acc[q][3]);
                *reinterpret_cast<float4*>(out + rowbase + cbase) = v0;
            }
            if (cbase + 7 < GRID) {
                float4 v1 = make_float4(acc[q][4], acc[q][5], acc[q][6], acc[q][7]);
                *reinterpret_cast<float4*>(out + rowbase + cbase + 4) = v1;
            }
        }
    }
}

// ================= launch =================
extern "C" void kernel_launch(void* const* d_in, const int* in_sizes, int n_in,
                              void* d_out, int out_size, void* d_ws, size_t ws_size,
                              hipStream_t stream) {
    const float* extrinsics = (const float*)d_in[0];
    const float* angles     = (const float*)d_in[1];
    const float* depth      = (const float*)d_in[2];
    const float* features   = (const float*)d_in[3];
    const float* opacities  = (const float*)d_in[4];
    float* out = (float*)d_out;

    float* ws      = (float*)d_ws;
    float* extinv  = ws + OFF_EXT;
    float* params  = ws + OFF_PAR;
    int*   tileCnt = (int*)ws + OFF_CNT;
    int*   tileOff = (int*)ws + OFF_OFFS;
    int*   tileCur = (int*)ws + OFF_CUR;
    int*   tileLst = (int*)ws + OFF_LIST;

    ext_inv_kernel<<<1, 16, 0, stream>>>(extrinsics, extinv);
    zero_cnt_kernel<<<(NTILES + 255) / 256, 256, 0, stream>>>(tileCnt);
    prep_kernel<<<(BN * NPIX + 255) / 256, 256, 0, stream>>>(
        angles, depth, opacities, extinv, params, tileCnt);
    scan_kernel<<<1, 32, 0, stream>>>(tileCnt, tileOff, tileCur);
    fill_kernel<<<(BN * NPIX + 255) / 256, 256, 0, stream>>>(params, tileCur, tileLst);
    splat_kernel<<<NTILES, 512, 0, stream>>>(features, params, tileOff, tileLst, out);
    (void)in_sizes; (void)n_in; (void)out_size; (void)ws_size;
}
